// LLaMAAttention_32684701122782
// MI455X (gfx1250) — matrix operand order repaired and checked by reading, not compiled, not run on hardware
//
#include <hip/hip_runtime.h>
#include <hip/hip_bf16.h>
#include <math.h>

// ---------------- problem constants ----------------
#define BATCH   2
#define SEQ     1024
#define HID     4096
#define NHEADS  32
#define NKVH    8
#define HD      128
#define GQA_G   (NHEADS / NKVH)     // 4
#define KVDIM   (NKVH * HD)         // 1024
#define MROWS   (BATCH * SEQ)       // 2048

typedef __bf16 bf16;
typedef bf16  v16bf __attribute__((ext_vector_type(16)));
typedef float v8f   __attribute__((ext_vector_type(8)));
typedef int   v4i   __attribute__((ext_vector_type(4)));

#define AS1 __attribute__((address_space(1)))
#define AS3 __attribute__((address_space(3)))

union FragU {            // 32 bytes = one 16-element bf16 WMMA fragment
    uint4 u[2];
    v16bf v;
};

static __device__ __forceinline__ int imin(int a, int b) { return a < b ? a : b; }

#if __has_builtin(__builtin_amdgcn_global_load_async_to_lds_b128)
#define HAS_ASYNC_LDS 1
#endif

// =====================================================================
// f32 -> bf16, 8 elements per thread, b128 in / b128 out
// =====================================================================
__global__ void cvt_bf16x8(const float* __restrict__ x, bf16* __restrict__ out, size_t n8) {
    size_t i = (size_t)blockIdx.x * blockDim.x + threadIdx.x;
    if (i >= n8) return;
    const float4* p = (const float4*)x + i * 2;
    float4 a = p[0], b = p[1];
    bf16 t[8] = {(bf16)a.x, (bf16)a.y, (bf16)a.z, (bf16)a.w,
                 (bf16)b.x, (bf16)b.y, (bf16)b.z, (bf16)b.w};
    ((uint4*)out)[i] = *(uint4*)t;
}

// =====================================================================
// GEMM: C[M,N](f32) = A[M,K](bf16,rowmajor) x B[K,N](bf16,rowmajor)
// Block tile 64x128, K-step 32.  A staged via async global->LDS b128,
// B staged transposed (padded stride 40 => conflict-free b128 reads).
// 8 waves: wave (wm 0..3, wn 0..1) owns 16 rows x 64 cols (4 WMMA/k-step).
// grid = (M/64, N/128), block = 256.
// =====================================================================
__global__ __launch_bounds__(256) void gemm_bf16_lds(const bf16* __restrict__ A,
                                                     const bf16* __restrict__ Bm,
                                                     float* __restrict__ C,
                                                     int M, int N, int K) {
    __shared__ __align__(16) bf16 Alds[64][40];    // 5.0 KB (padded)
    __shared__ __align__(16) bf16 Blds[128][40];   // 10.0 KB, transposed: [n][k]

    const int t    = threadIdx.x;
    const int lane = t & 31;
    const int wave = t >> 5;
    const int wm   = wave & 3;          // 4 M sub-tiles
    const int wn   = wave >> 2;         // 2 N sub-strips of 64
    const int m0   = blockIdx.x * 64;
    const int n0   = blockIdx.y * 128;
    const int cn   = lane & 15;
    const int abase = (lane < 16) ? 0 : 16;    // A K-split per lane half
    const int bbase = (lane < 16) ? 0 : 16;    // B K-split per lane half

    // staging roles
    const int arow = t >> 2;            // 0..63
    const int achk = (t & 3) * 8;       // 0,8,16,24
    const int brow = t >> 3;            // 0..31   (k within step)
    const int bseg = (t & 7) * 16;      // 0..112  (n within tile)

    v8f acc[4] = {};

    for (int k0 = 0; k0 < K; k0 += 32) {
        // ---- stage A tile 64x32: one async b128 per thread ----
        const bf16* gA = &A[(size_t)(m0 + arow) * K + k0 + achk];
#ifdef HAS_ASYNC_LDS
        __builtin_amdgcn_global_load_async_to_lds_b128(
            (AS1 v4i*)(const void*)gA,
            (AS3 v4i*)(void*)&Alds[arow][achk], 0, 0);
#else
        *(uint4*)&Alds[arow][achk] = *(const uint4*)gA;
#endif
        // ---- stage B tile 32x128 transposed: 32B load + 16 b16 scatter ----
        {
            const bf16* gB = &Bm[(size_t)(k0 + brow) * N + n0 + bseg];
            bf16 tmp[16];
            *(uint4*)tmp       = *(const uint4*)gB;
            *(uint4*)(tmp + 8) = *(const uint4*)(gB + 8);
#pragma unroll
            for (int j = 0; j < 16; ++j)
                Blds[bseg + j][brow] = tmp[j];
        }
#ifdef HAS_ASYNC_LDS
#if __has_builtin(__builtin_amdgcn_s_wait_asynccnt)
        __builtin_amdgcn_s_wait_asynccnt(0);
#endif
#endif
        __syncthreads();

        // ---- compute: 1 A-frag + 4 B-frags, 4 WMMA ----
        FragU af;
        af.u[0] = *(const uint4*)&Alds[wm * 16 + cn][abase];       // K = abase..abase+7
        af.u[1] = *(const uint4*)&Alds[wm * 16 + cn][abase + 8];   // K = abase+8..+15
#pragma unroll
        for (int tt = 0; tt < 4; ++tt) {
            const int n = wn * 64 + tt * 16 + cn;
            FragU bfr;
            bfr.u[0] = *(const uint4*)&Blds[n][bbase];       // K = bbase..bbase+7
            bfr.u[1] = *(const uint4*)&Blds[n][bbase + 8];   // K = bbase+8..+15
            acc[tt] = __builtin_amdgcn_wmma_f32_16x16x32_bf16(
                false, af.v, false, bfr.v, (short)0, acc[tt], false, false);
        }
        __syncthreads();
    }

#pragma unroll
    for (int tt = 0; tt < 4; ++tt) {
        const int col = n0 + wn * 64 + tt * 16 + cn;
#pragma unroll
        for (int r = 0; r < 8; ++r) {
            int m = m0 + wm * 16 + r + 8 * (lane >> 4);
            C[(size_t)m * N + col] = acc[tt][r];
        }
    }
}

// =====================================================================
// RoPE (rotate_half, cat(freqs,freqs)) + f32 -> bf16
// =====================================================================
__global__ void rope_cvt_bf16(const float* __restrict__ x,
                              const int* __restrict__ pos_ids,
                              bf16* __restrict__ out,
                              int nheads, size_t total) {
    size_t idx = (size_t)blockIdx.x * blockDim.x + threadIdx.x;
    if (idx >= total) return;
    int d       = (int)(idx & 63);
    size_t rest = idx >> 6;
    int h       = (int)(rest % nheads);
    size_t bs   = rest / nheads;
    int s       = (int)(bs % SEQ);

    float pos  = (float)pos_ids[s];
    float invf = __expf(-((float)(2 * d) / (float)HD) * logf(10000.0f));
    float ang  = pos * invf;
    float c, sn;
    __sincosf(ang, &sn, &c);

    size_t base = (bs * (size_t)nheads + h) * HD;
    float x0 = x[base + d];
    float x1 = x[base + d + 64];
    out[base + d]      = (bf16)(x0 * c - x1 * sn);
    out[base + d + 64] = (bf16)(x1 * c + x0 * sn);
}

// =====================================================================
// V: f32 [B*S, NKV*HD] -> bf16 transposed [B, NKV, HD, S]  (8 s per thread)
// =====================================================================
__global__ void cvt_v_transpose(const float* __restrict__ vf, bf16* __restrict__ vbT) {
    size_t idx = (size_t)blockIdx.x * blockDim.x + threadIdx.x;   // B*NKV*HD*(S/8)
    if (idx >= (size_t)BATCH * NKVH * HD * (SEQ / 8)) return;
    int s8      = (int)(idx % (SEQ / 8));
    size_t rest = idx / (SEQ / 8);
    int hd      = (int)(rest % HD);
    rest /= HD;
    int kv = (int)(rest % NKVH);
    int b  = (int)(rest / NKVH);

    bf16 tmp[8];
#pragma unroll
    for (int j = 0; j < 8; ++j)
        tmp[j] = (bf16)vf[((size_t)(b * SEQ + s8 * 8 + j) * NKVH + kv) * HD + hd];
    *(uint4*)&vbT[((size_t)(b * NKVH + kv) * HD + hd) * SEQ + s8 * 8] = *(uint4*)tmp;
}

// =====================================================================
// Flash-style causal GQA attention (bf16 in, bf16 ctx out).
// One wave per 16-query tile; Q resident as 4 A-fragments (2xb128 loads);
// K-frags contiguous b128 loads; P staged via padded LDS; V-frags from
// transposed V, contiguous b128 loads.
// grid = (SEQ/64, NHEADS, BATCH), block = 128 (4 waves).
// =====================================================================
__global__ __launch_bounds__(128) void attn_wmma(const bf16* __restrict__ qb,
                                                 const bf16* __restrict__ kb,
                                                 const bf16* __restrict__ vbT,
                                                 bf16* __restrict__ ctx) {
    const int lane  = threadIdx.x & 31;
    const int wave  = threadIdx.x >> 5;
    const int qtile = blockIdx.x * 4 + wave;
    const int head  = blockIdx.y;
    const int b     = blockIdx.z;
    const int kv    = head / GQA_G;
    const int cn    = lane & 15;
    const int hi    = lane >> 4;
    const int abase = (lane < 16) ? 0 : 16;
    const int bbase = (lane < 16) ? 0 : 16;

    __shared__ __align__(16) bf16 plds[4][16][40];   // padded: conflict-free b128

    // ---- resident Q fragments: 16x128 = 4 x (16x32) ----
    const size_t qrowbase = ((size_t)(b * SEQ + qtile * 16 + cn) * NHEADS + head) * HD;
    FragU qa[4];
#pragma unroll
    for (int ks = 0; ks < 4; ++ks) {
        qa[ks].u[0] = *(const uint4*)&qb[qrowbase + ks * 32 + abase];
        qa[ks].u[1] = *(const uint4*)&qb[qrowbase + ks * 32 + abase + 8];
    }

    v8f acc[8] = {};
    float mrow[8], lrow[8];
#pragma unroll
    for (int r = 0; r < 8; ++r) { mrow[r] = -1e30f; lrow[r] = 0.0f; }

    const float scale = 0.08838834764831845f;   // 1/sqrt(128)
    const size_t vhead = ((size_t)(b * NKVH + kv) * HD) * SEQ;   // vbT head base

    for (int kbase = 0; kbase <= qtile * 16; kbase += 32) {
        // ---------- scores: two 16x16 subtiles (K = HD = 128) ----------
        v8f sc[2];
#pragma unroll
        for (int sub = 0; sub < 2; ++sub) {
            v8f s = {};
            const int key = imin(kbase + sub * 16 + cn, SEQ - 1);
            const size_t krow = ((size_t)(b * SEQ + key) * NKVH + kv) * HD;
#pragma unroll
            for (int ks = 0; ks < 4; ++ks) {
                FragU bfr;
                bfr.u[0] = *(const uint4*)&kb[krow + ks * 32 + bbase];
                bfr.u[1] = *(const uint4*)&kb[krow + ks * 32 + bbase + 8];
                s = __builtin_amdgcn_wmma_f32_16x16x32_bf16(
                    false, qa[ks].v, false, bfr.v, (short)0, s, false, false);
            }
            sc[sub] = s;
        }

        // ---------- causal mask + online softmax ----------
#pragma unroll
        for (int r = 0; r < 8; ++r) {
            const int qr   = qtile * 16 + r + 8 * hi;
            const int key0 = kbase + cn;
            const int key1 = kbase + 16 + cn;
            float s0 = sc[0][r] * scale; if (key0 > qr) s0 = -1e30f;
            float s1 = sc[1][r] * scale; if (key1 > qr) s1 = -1e30f;

            float v = fmaxf(s0, s1);
#pragma unroll
            for (int off = 8; off >= 1; off >>= 1)
                v = fmaxf(v, __shfl_xor(v, off, 32));

            const float mn    = fmaxf(mrow[r], v);
            const float alpha = __expf(mrow[r] - mn);
            mrow[r] = mn;

            const float p0 = __expf(s0 - mn);
            const float p1 = __expf(s1 - mn);
            plds[wave][r + 8 * hi][cn]      = (bf16)p0;
            plds[wave][r + 8 * hi][cn + 16] = (bf16)p1;

            float tsum = p0 + p1;
#pragma unroll
            for (int off = 8; off >= 1; off >>= 1)
                tsum += __shfl_xor(tsum, off, 32);
            lrow[r] = lrow[r] * alpha + tsum;

#pragma unroll
            for (int tt = 0; tt < 8; ++tt) acc[tt][r] *= alpha;
        }

        // ---------- P A-fragment from LDS (2 x ds_load_b128) ----------
        FragU pa;
        pa.u[0] = *(const uint4*)&plds[wave][cn][abase];
        pa.u[1] = *(const uint4*)&plds[wave][cn][abase + 8];

        // ---------- O += P(16x32) * V(32x128): 8 WMMAs ----------
#pragma unroll
        for (int ht = 0; ht < 8; ++ht) {
            const size_t vcol = vhead + (size_t)(ht * 16 + cn) * SEQ + kbase + bbase;
            FragU bfr;
            bfr.u[0] = *(const uint4*)&vbT[vcol];
            bfr.u[1] = *(const uint4*)&vbT[vcol + 8];
            acc[ht] = __builtin_amdgcn_wmma_f32_16x16x32_bf16(
                false, pa.v, false, bfr.v, (short)0, acc[ht], false, false);
        }
    }

    // ---------- normalize & store ctx [B,S,NH,HD] as bf16 ----------
#pragma unroll
    for (int ht = 0; ht < 8; ++ht)
#pragma unroll
        for (int r = 0; r < 8; ++r) {
            const int qr = qtile * 16 + r + 8 * hi;
            float o = acc[ht][r] / lrow[r];
            ctx[((size_t)(b * SEQ + qr) * NHEADS + head) * HD + ht * 16 + cn] = (bf16)o;
        }
}

// =====================================================================
// launch
// =====================================================================
extern "C" void kernel_launch(void* const* d_in, const int* in_sizes, int n_in,
                              void* d_out, int out_size, void* d_ws, size_t ws_size,
                              hipStream_t stream) {
    const float* hs  = (const float*)d_in[0];
    /* d_in[1] = attention_mask: causal, computed analytically in-kernel */
    const int*   pos = (const int*)d_in[2];
    const float* Wq  = (const float*)d_in[3];
    const float* Wk  = (const float*)d_in[4];
    const float* Wv  = (const float*)d_in[5];
    const float* Wo  = (const float*)d_in[6];
    float* out = (float*)d_out;

    char* ws = (char*)d_ws;
    bf16*  hsb  = (bf16*)ws;  ws += (size_t)MROWS * HID   * sizeof(bf16);    // 16 MiB
    bf16*  wb   = (bf16*)ws;  ws += (size_t)HID   * HID   * sizeof(bf16);    // 32 MiB (reused)
    float* qf   = (float*)ws; ws += (size_t)MROWS * HID   * sizeof(float);   // 32 MiB
    float* kf   = (float*)ws; ws += (size_t)MROWS * KVDIM * sizeof(float);   //  8 MiB
    float* vf   = (float*)ws; ws += (size_t)MROWS * KVDIM * sizeof(float);   //  8 MiB
    bf16*  qbb  = (bf16*)ws;  ws += (size_t)MROWS * HID   * sizeof(bf16);    // 16 MiB
    bf16*  kbb  = (bf16*)ws;  ws += (size_t)MROWS * KVDIM * sizeof(bf16);    //  4 MiB
    bf16*  vbT  = (bf16*)ws;  ws += (size_t)MROWS * KVDIM * sizeof(bf16) + 128; // 4 MiB + OOB slack
    bf16*  ctxb = (bf16*)ws;  ws += (size_t)MROWS * HID   * sizeof(bf16);    // 16 MiB

    dim3 blk(256);
    auto nblk = [](size_t n) { return dim3((unsigned)((n + 255) / 256)); };

    // hidden states -> bf16
    cvt_bf16x8<<<nblk((size_t)MROWS * HID / 8), blk, 0, stream>>>(hs, hsb, (size_t)MROWS * HID / 8);

    // Q projection
    cvt_bf16x8<<<nblk((size_t)HID * HID / 8), blk, 0, stream>>>(Wq, wb, (size_t)HID * HID / 8);
    gemm_bf16_lds<<<dim3(MROWS / 64, HID / 128), blk, 0, stream>>>(hsb, wb, qf, MROWS, HID, HID);
    // K projection
    cvt_bf16x8<<<nblk((size_t)HID * KVDIM / 8), blk, 0, stream>>>(Wk, wb, (size_t)HID * KVDIM / 8);
    gemm_bf16_lds<<<dim3(MROWS / 64, KVDIM / 128), blk, 0, stream>>>(hsb, wb, kf, MROWS, KVDIM, HID);
    // V projection
    cvt_bf16x8<<<nblk((size_t)HID * KVDIM / 8), blk, 0, stream>>>(Wv, wb, (size_t)HID * KVDIM / 8);
    gemm_bf16_lds<<<dim3(MROWS / 64, KVDIM / 128), blk, 0, stream>>>(hsb, wb, vf, MROWS, KVDIM, HID);

    // RoPE + bf16 convert; V transpose
    {
        size_t nq = (size_t)MROWS * NHEADS * 64;
        rope_cvt_bf16<<<nblk(nq), blk, 0, stream>>>(qf, pos, qbb, NHEADS, nq);
        size_t nk = (size_t)MROWS * NKVH * 64;
        rope_cvt_bf16<<<nblk(nk), blk, 0, stream>>>(kf, pos, kbb, NKVH, nk);
        size_t nv = (size_t)BATCH * NKVH * HD * (SEQ / 8);
        cvt_v_transpose<<<nblk(nv), blk, 0, stream>>>(vf, vbT);
    }

    // causal GQA attention -> ctx (bf16)
    attn_wmma<<<dim3(SEQ / 64, NHEADS, BATCH), dim3(128), 0, stream>>>(qbb, kbb, vbT, ctxb);

    // output projection
    cvt_bf16x8<<<nblk((size_t)HID * HID / 8), blk, 0, stream>>>(Wo, wb, (size_t)HID * HID / 8);
    gemm_bf16_lds<<<dim3(MROWS / 64, HID / 128), blk, 0, stream>>>(ctxb, wb, out, MROWS, HID, HID);
}
